// CUTLASSTransformerBlock_5686536700375
// MI455X (gfx1250) — matrix operand order repaired and checked by reading, not compiled, not run on hardware
//
#include <hip/hip_runtime.h>
#include <stdint.h>

// ---------------------------------------------------------------------------
// Types for CDNA5 WMMA (wave32): v_wmma_f32_16x16x32_bf16
// ---------------------------------------------------------------------------
typedef __attribute__((ext_vector_type(16))) __bf16 v16bf;
typedef __attribute__((ext_vector_type(8)))  float  v8f;
typedef unsigned short u16;
typedef unsigned int   u32;
typedef __attribute__((ext_vector_type(4))) unsigned int u32x4;
typedef __attribute__((ext_vector_type(4))) int          i32x4;
typedef __attribute__((ext_vector_type(8))) int          i32x8;

union Frag {
    v16bf v;
    uint4 q[2];
    u16   h[16];
};

__device__ inline u16 f2bf(float f) {
    uint32_t u = __float_as_uint(f);
    uint32_t r = u + 0x7FFFu + ((u >> 16) & 1u);   // round-to-nearest-even
    return (u16)(r >> 16);
}

__device__ inline float gelu_exact(float x) {
    return 0.5f * x * (1.0f + erff(x * 0.70710678118654752f));
}

__device__ inline v8f wmma_bf16(const Frag& a, const Frag& b, v8f c) {
    return __builtin_amdgcn_wmma_f32_16x16x32_bf16(
        false, a.v, false, b.v, (short)0, c, false, false);
}

// ---------------------------------------------------------------------------
// DPP-based 16-lane butterfly reductions (inside each 16-lane DPP row =
// C-fragment lane-half).  No LDS traffic.
// ---------------------------------------------------------------------------
#if defined(__has_builtin)
#if __has_builtin(__builtin_amdgcn_update_dpp)
#define HAVE_DPP 1
#endif
#endif

#ifdef HAVE_DPP
#define DPP_STEP(X, CTRL, OPC)                                                 \
    {                                                                          \
        int _y = __builtin_amdgcn_update_dpp(0, __float_as_int(X), CTRL, 0xF,  \
                                             0xF, true);                       \
        X = OPC(X, __int_as_float(_y));                                        \
    }
__device__ inline float row16_max(float x) {
    DPP_STEP(x, 0xB1, fmaxf)   // quad_perm [1,0,3,2]  (xor 1)
    DPP_STEP(x, 0x4E, fmaxf)   // quad_perm [2,3,0,1]  (xor 2)
    DPP_STEP(x, 0x141, fmaxf)  // row_half_mirror      (xor 4 equiv.)
    DPP_STEP(x, 0x140, fmaxf)  // row_mirror           (xor 8 equiv.)
    return x;
}
__device__ inline float fadd2(float a, float b) { return a + b; }
__device__ inline float row16_sum(float x) {
    DPP_STEP(x, 0xB1, fadd2)
    DPP_STEP(x, 0x4E, fadd2)
    DPP_STEP(x, 0x141, fadd2)
    DPP_STEP(x, 0x140, fadd2)
    return x;
}
#else
__device__ inline float row16_max(float x) {
#pragma unroll
    for (int off = 1; off < 16; off <<= 1) x = fmaxf(x, __shfl_xor(x, off, 32));
    return x;
}
__device__ inline float row16_sum(float x) {
#pragma unroll
    for (int off = 1; off < 16; off <<= 1) x += __shfl_xor(x, off, 32);
    return x;
}
#endif

// ---------------------------------------------------------------------------
// Tensor Data Mover: async 2-D tile load global->LDS with LDS row padding.
// D# per CDNA5 ISA 8.3/8.4 (data_size=2B).  Issued by one wave; completion
// ordered on that wave's TENSORcnt (in-order), published via barrier.
// ---------------------------------------------------------------------------
#if defined(__has_builtin)
#if __has_builtin(__builtin_amdgcn_tensor_load_to_lds) && \
    __has_builtin(__builtin_amdgcn_s_wait_tensorcnt)
#define HAVE_TDM 1
#endif
#endif

#ifdef HAVE_TDM
// tile0 = elements/row, tile1 = rows, stride0 = row stride (elements).
// pad_int/pad_amt: D# encodings (interval 2^(v+1) DWORDs, pad amount v+1 DW).
__device__ inline void tdm_load_2d(u32 lds_addr, const void* gptr,
                                   u32 tile0, u32 tile1, u32 stride0,
                                   u32 pad_int, u32 pad_amt) {
    uint64_t ga = (uint64_t)gptr;
    u32x4 g0;
    g0.x = 1u;                                   // count=1, user descriptor
    g0.y = lds_addr;                             // LDS byte address
    g0.z = (u32)ga;                              // global_addr[31:0]
    g0.w = (u32)(ga >> 32) | (2u << 30);         // global_addr[56:32] | type=2
    i32x8 g1;
    g1[0] = (int)((1u << 16) |                   // data_size = 2 bytes
                  (1u << 20) |                   // pad_enable
                  (pad_int << 22) | (pad_amt << 25));
    g1[1] = (int)(tile0 << 16);                  // tensor_dim0[15:0] (=tile0)
    g1[2] = (int)(tile1 << 16);                  // tensor_dim1[15:0] (=tile1)
    g1[3] = (int)(tile0 << 16);                  // tile_dim0
    g1[4] = (int)(tile1 & 0xFFFFu);              // tile_dim1 (tile_dim2 = 0)
    g1[5] = (int)stride0;                        // tensor_dim0_stride[31:0]
    g1[6] = 0;
    g1[7] = 0;
    i32x4 z4 = {0, 0, 0, 0};
#if __clang_major__ >= 23
    i32x8 z8 = {0, 0, 0, 0, 0, 0, 0, 0};
    __builtin_amdgcn_tensor_load_to_lds(g0, g1, z4, z4, z8, 0);
#else
    __builtin_amdgcn_tensor_load_to_lds(g0, g1, z4, z4, 0);
#endif
}
#endif

// ---------------------------------------------------------------------------
// fp32 -> bf16 conversion (weights)
// ---------------------------------------------------------------------------
__global__ void f32_to_bf16_kernel(const float* __restrict__ src,
                                   u16* __restrict__ dst, int n) {
    int i = blockIdx.x * blockDim.x + threadIdx.x;
    if (i < n) dst[i] = f2bf(src[i]);
}

// ---------------------------------------------------------------------------
// RMSNorm: one block per row of [rows, 1024]; fp32 in, bf16 out
// ---------------------------------------------------------------------------
__global__ __launch_bounds__(256)
void rmsnorm_bf16_kernel(const float* __restrict__ x,
                         const float* __restrict__ w,
                         u16* __restrict__ out, int D) {
    const int row = blockIdx.x;
    const int tid = threadIdx.x;
    const float* xr = x + (size_t)row * D;

    float v0[4];
    float ss = 0.f;
#pragma unroll
    for (int i = 0; i < 4; ++i) {
        v0[i] = xr[tid + i * 256];
        ss += v0[i] * v0[i];
    }
#pragma unroll
    for (int off = 16; off >= 1; off >>= 1) ss += __shfl_xor(ss, off, 32);
    __shared__ float red[8];
    if ((tid & 31) == 0) red[tid >> 5] = ss;
    __syncthreads();
    float tot = 0.f;
#pragma unroll
    for (int i = 0; i < 8; ++i) tot += red[i];

    const float scale = rsqrtf(tot / (float)D + 1e-6f);
    u16* orow = out + (size_t)row * D;
#pragma unroll
    for (int i = 0; i < 4; ++i) {
        int idx = tid + i * 256;
        orow[idx] = f2bf(v0[i] * scale * w[idx]);
    }
}

// ---------------------------------------------------------------------------
// Generic bf16 WMMA GEMM: C[M,N] = A[M,K] * B[K,N] + bias
//   BM=128 BN=128 BK=32, 256 threads = 8 waves (2x4), wave tile 64x32.
//   Double-buffered: A tiles streamed by TDM one step ahead (TENSORcnt<=1
//   overlap), B tiles prefetched into registers one step ahead.
//   MODE 0: fp32 out   MODE 1: bf16 out (+GELU)   MODE 2: QKV scatter
// ---------------------------------------------------------------------------
template <int MODE, bool GELU>
__global__ __launch_bounds__(256)
void gemm_bf16_kernel(const u16* __restrict__ A, const u16* __restrict__ B,
                      const float* __restrict__ bias,
                      float* __restrict__ outF, u16* __restrict__ outH,
                      u16* __restrict__ qOut, u16* __restrict__ kOut,
                      u16* __restrict__ vOut,
                      int M, int N, int K) {
    __shared__ __align__(16) u16 As[2][128][40];   // [m][k], stride 40
    __shared__ __align__(16) u16 Bt[2][128][40];   // [n][k], B transposed

    const int tid  = threadIdx.x;
    const int wid  = tid >> 5, lane = tid & 31;
    const int hf   = lane >> 4, lm = lane & 15;
    const int wm   = wid >> 2, wn = wid & 3;           // waves 2x4
    const int m0   = blockIdx.y * 128, n0 = blockIdx.x * 128;
    const int nk   = K >> 5;

    v8f c[4][2];
#pragma unroll
    for (int mt = 0; mt < 4; ++mt)
#pragma unroll
        for (int nt = 0; nt < 2; ++nt)
#pragma unroll
            for (int r = 0; r < 8; ++r) c[mt][nt][r] = 0.f;

    const int lb_r = tid >> 3, lb_c = (tid & 7) * 16;   // B tile: 32x128
#ifndef HAVE_TDM
    const int la_r = tid >> 1, la_c = (tid & 1) * 16;   // fallback A loader
#endif

    // prologue: B tile 0 -> registers; TDM for A tile 0
    uint4 bpre0, bpre1;
    {
        const uint4* src = (const uint4*)(B + (size_t)lb_r * N + n0 + lb_c);
        bpre0 = src[0];
        bpre1 = src[1];
    }
#ifdef HAVE_TDM
    if (wid == 0) {
        // 128 rows x 32 elems: row = 16 DWORDs, pad 4 DWORDs -> stride 40
        tdm_load_2d((u32)(uintptr_t)&As[0][0][0], A + (size_t)m0 * K,
                    32u, 128u, (u32)K, 3u, 3u);
    }
#endif

    for (int kt = 0; kt < nk; ++kt) {
        const int buf = kt & 1;
        {   // commit prefetched B tile -> LDS transposed [n][k]
            __align__(16) u16 tmp[16];
            *(uint4*)&tmp[0] = bpre0;
            *(uint4*)&tmp[8] = bpre1;
#pragma unroll
            for (int j = 0; j < 16; ++j) Bt[buf][lb_c + j][lb_r] = tmp[j];
        }
        const bool next = (kt + 1 < nk);
        if (next) {   // prefetch next B tile into registers (overlaps compute)
            const uint4* src =
                (const uint4*)(B + (size_t)((kt + 1) * 32 + lb_r) * N + n0 + lb_c);
            bpre0 = src[0];
            bpre1 = src[1];
        }
#ifdef HAVE_TDM
        if (wid == 0) {
            if (next)   // stream next A tile into the other buffer
                tdm_load_2d((u32)(uintptr_t)&As[buf ^ 1][0][0],
                            A + (size_t)m0 * K + (kt + 1) * 32,
                            32u, 128u, (u32)K, 3u, 3u);
            // TENSORcnt is in-order: <=1 outstanding => tile kt complete
            if (next) __builtin_amdgcn_s_wait_tensorcnt(1);
            else      __builtin_amdgcn_s_wait_tensorcnt(0);
        }
#else
        {
            const uint4* src =
                (const uint4*)(A + (size_t)(m0 + la_r) * K + kt * 32 + la_c);
            uint4 a0 = src[0], a1 = src[1];
            *(uint4*)&As[buf][la_r][la_c]     = a0;
            *(uint4*)&As[buf][la_r][la_c + 8] = a1;
        }
#endif
        __syncthreads();

        Frag a[4], b[2];
#pragma unroll
        for (int mt = 0; mt < 4; ++mt) {
            // A frag: elem e<8 -> K = hf*8+e ; e>=8 -> K = 16+hf*8+(e-8)
            int row = wm * 64 + mt * 16 + lm;
            int kl  = hf * 8;
            a[mt].q[0] = *(const uint4*)&As[buf][row][kl];
            a[mt].q[1] = *(const uint4*)&As[buf][row][kl + 16];
        }
#pragma unroll
        for (int nt = 0; nt < 2; ++nt) {
            // B frag: elem e<8 -> K = hf*8+e ; e>=8 -> K = 16+hf*8+(e-8), N = lm
            int nn   = wn * 32 + nt * 16 + lm;
            int koff = hf * 8;
            b[nt].q[0] = *(const uint4*)&Bt[buf][nn][koff];
            b[nt].q[1] = *(const uint4*)&Bt[buf][nn][koff + 16];
        }
#pragma unroll
        for (int mt = 0; mt < 4; ++mt)
#pragma unroll
            for (int nt = 0; nt < 2; ++nt)
                c[mt][nt] = wmma_bf16(a[mt], b[nt], c[mt][nt]);
        __syncthreads();   // reads done before next iteration's writes
    }

    // Epilogue. C layout: vgpr r, lanes 0-15: M=r, N=lane; lanes 16-31: M=8+r.
#pragma unroll
    for (int mt = 0; mt < 4; ++mt)
#pragma unroll
        for (int nt = 0; nt < 2; ++nt)
#pragma unroll
            for (int r = 0; r < 8; ++r) {
                int m = m0 + wm * 64 + mt * 16 + r + 8 * hf;
                int n = n0 + wn * 32 + nt * 16 + lm;
                float val = c[mt][nt][r] + bias[n];
                if (GELU) val = gelu_exact(val);
                if (MODE == 0) {
                    outF[(size_t)m * N + n] = val;
                } else if (MODE == 1) {
                    outH[(size_t)m * N + n] = f2bf(val);
                } else {
                    int which = n >> 10, hh = (n >> 6) & 15, d = n & 63;
                    int bi = m >> 11, s = m & 2047;
                    u16* dst = (which == 0) ? qOut : (which == 1) ? kOut : vOut;
                    dst[((((size_t)bi * 16 + hh) * 2048 + s) << 6) + d] = f2bf(val);
                }
            }
}

// ---------------------------------------------------------------------------
// Flash attention: q,k,v in [B,H,S,64] bf16 -> ctx [B,S,H*64] bf16.
// Block = 128 threads (4 waves) = 64 query rows of one (b,h); keys streamed
// in tiles of 32.  K tiles double-buffered via TDM, V prefetched in regs and
// stored transposed, DPP softmax reductions.
// ---------------------------------------------------------------------------
__global__ __launch_bounds__(128)
void flash_attn_kernel(const u16* __restrict__ Q, const u16* __restrict__ Km,
                       const u16* __restrict__ Vm, u16* __restrict__ ctx) {
    __shared__ __align__(16) u16 qs[64][72];        // Q tile  [q][d]
    __shared__ __align__(16) u16 ks[2][32][72];     // K tiles [key][d]
    __shared__ __align__(16) u16 vts[2][64][40];    // V^T tiles [d][key]
    __shared__ __align__(16) u16 pa[4][16][40];     // per-wave P (16x32) bf16

    const int tid  = threadIdx.x;
    const int wid  = tid >> 5, lane = tid & 31;
    const int hf   = lane >> 4, lm = lane & 15;
    const int qblk = blockIdx.x, head = blockIdx.y, batch = blockIdx.z;

    const size_t hb = ((size_t)batch * 16 + head) * 2048;
    const u16* qp = Q  + (hb + (size_t)qblk * 64) * 64;
    const u16* kp = Km + hb * 64;
    const u16* vp = Vm + hb * 64;

    const int lk_r = tid >> 2, lk_c = (tid & 3) * 16;

    // prologue: V tile 0 -> registers; Q and K tile 0 via TDM
    uint4 vpre0, vpre1;
    {
        const uint4* src = (const uint4*)(vp + (size_t)lk_r * 64 + lk_c);
        vpre0 = src[0];
        vpre1 = src[1];
    }
#ifdef HAVE_TDM
    if (wid == 0) {
        // rows of 64 elems = 32 DWORDs, pad 4 DWORDs -> stride 72
        tdm_load_2d((u32)(uintptr_t)&qs[0][0], qp, 64u, 64u, 64u, 4u, 3u);
        tdm_load_2d((u32)(uintptr_t)&ks[0][0][0], kp, 64u, 32u, 64u, 4u, 3u);
        __builtin_amdgcn_s_wait_tensorcnt(1);   // in-order => Q complete
    }
#else
    {
        int r = tid >> 1, cc = (tid & 1) * 32;
        const uint4* src = (const uint4*)(qp + r * 64 + cc);
#pragma unroll
        for (int j = 0; j < 4; ++j) *(uint4*)&qs[r][cc + j * 8] = src[j];
    }
#endif
    __syncthreads();

    Frag aq[2];
#pragma unroll
    for (int dc = 0; dc < 2; ++dc) {
        int row = wid * 16 + lm;
        int kl  = dc * 32 + hf * 8;
        aq[dc].q[0] = *(const uint4*)&qs[row][kl];
        aq[dc].q[1] = *(const uint4*)&qs[row][kl + 16];
    }

    float mrow[8], lrow[8];
    v8f o[4];
#pragma unroll
    for (int r = 0; r < 8; ++r) { mrow[r] = -1e30f; lrow[r] = 0.f; }
#pragma unroll
    for (int t = 0; t < 4; ++t)
#pragma unroll
        for (int r = 0; r < 8; ++r) o[t][r] = 0.f;

    for (int kt = 0; kt < 64; ++kt) {          // 2048 keys / 32
        const int buf = kt & 1;
        {   // commit prefetched V tile -> LDS transposed [d][key]
            __align__(16) u16 tmp[16];
            *(uint4*)&tmp[0] = vpre0;
            *(uint4*)&tmp[8] = vpre1;
#pragma unroll
            for (int j = 0; j < 16; ++j) vts[buf][lk_c + j][lk_r] = tmp[j];
        }
        const bool next = (kt + 1 < 64);
        if (next) {   // prefetch next V tile into registers
            const uint4* src =
                (const uint4*)(vp + (size_t)((kt + 1) * 32 + lk_r) * 64 + lk_c);
            vpre0 = src[0];
            vpre1 = src[1];
        }
#ifdef HAVE_TDM
        if (wid == 0) {
            if (next)
                tdm_load_2d((u32)(uintptr_t)&ks[buf ^ 1][0][0],
                            kp + (size_t)((kt + 1) * 32) * 64, 64u, 32u, 64u,
                            4u, 3u);
            if (next) __builtin_amdgcn_s_wait_tensorcnt(1);   // tile kt done
            else      __builtin_amdgcn_s_wait_tensorcnt(0);
        }
#else
        {
            const uint4* src =
                (const uint4*)(kp + (size_t)(kt * 32 + lk_r) * 64 + lk_c);
            *(uint4*)&ks[buf][lk_r][lk_c]     = src[0];
            *(uint4*)&ks[buf][lk_r][lk_c + 8] = src[1];
        }
#endif
        __syncthreads();

        // scores: S = Q * K^T   (two 16-key column tiles, two d-chunks)
        v8f s[2];
#pragma unroll
        for (int nt = 0; nt < 2; ++nt) {
#pragma unroll
            for (int r = 0; r < 8; ++r) s[nt][r] = 0.f;
#pragma unroll
            for (int dc = 0; dc < 2; ++dc) {
                Frag bk;
                int n    = nt * 16 + lm;
                int koff = dc * 32 + hf * 8;
                bk.q[0] = *(const uint4*)&ks[buf][n][koff];
                bk.q[1] = *(const uint4*)&ks[buf][n][koff + 16];
                s[nt] = wmma_bf16(aq[dc], bk, s[nt]);
            }
        }

        // online softmax; row stats uniform across the 16 lanes of a half
#pragma unroll
        for (int r = 0; r < 8; ++r) {
            float s0 = s[0][r] * 0.125f;       // 1/sqrt(64)
            float s1 = s[1][r] * 0.125f;
            float mx = row16_max(fmaxf(s0, s1));
            float mnew  = fmaxf(mrow[r], mx);
            float alpha = __expf(mrow[r] - mnew);
            float p0 = __expf(s0 - mnew), p1 = __expf(s1 - mnew);
            float sum = row16_sum(p0 + p1);
            lrow[r] = lrow[r] * alpha + sum;
            mrow[r] = mnew;
            int mrw = r + 8 * hf;
            pa[wid][mrw][lm]      = f2bf(p0);
            pa[wid][mrw][16 + lm] = f2bf(p1);
#pragma unroll
            for (int t = 0; t < 4; ++t) o[t][r] *= alpha;
        }

        // O += P * V  (per-wave DS ops stay in order; no barrier needed)
        Frag ap;
        ap.q[0] = *(const uint4*)&pa[wid][lm][hf * 8];
        ap.q[1] = *(const uint4*)&pa[wid][lm][hf * 8 + 16];
#pragma unroll
        for (int t = 0; t < 4; ++t) {
            Frag bv;
            int d    = t * 16 + lm;
            int koff = hf * 8;
            bv.q[0] = *(const uint4*)&vts[buf][d][koff];
            bv.q[1] = *(const uint4*)&vts[buf][d][koff + 16];
            o[t] = wmma_bf16(ap, bv, o[t]);
        }
        __syncthreads();   // reads done before next iteration's writes
    }

    // normalize and write ctx [B, S, H*64] as bf16
#pragma unroll
    for (int r = 0; r < 8; ++r) {
        float inv = 1.0f / lrow[r];
        int qg = qblk * 64 + wid * 16 + r + 8 * hf;
#pragma unroll
        for (int t = 0; t < 4; ++t) {
            int d = t * 16 + lm;
            ctx[(((size_t)batch * 2048 + qg) * 16 + head) * 64 + d] =
                f2bf(o[t][r] * inv);
        }
    }
}

// ---------------------------------------------------------------------------
// Host launch
// ---------------------------------------------------------------------------
extern "C" void kernel_launch(void* const* d_in, const int* in_sizes, int n_in,
                              void* d_out, int out_size, void* d_ws, size_t ws_size,
                              hipStream_t stream) {
    const float* x     = (const float*)d_in[0];
    const float* W_qkv = (const float*)d_in[1];
    const float* b_qkv = (const float*)d_in[2];
    const float* W_o   = (const float*)d_in[3];
    const float* b_o   = (const float*)d_in[4];
    const float* W_ff1 = (const float*)d_in[5];
    const float* b_ff1 = (const float*)d_in[6];
    const float* W_ff2 = (const float*)d_in[7];
    const float* b_ff2 = (const float*)d_in[8];
    const float* n1w   = (const float*)d_in[9];
    const float* n2w   = (const float*)d_in[10];
    float* out = (float*)d_out;

    char* ws = (char*)d_ws;
    u16*   wqkv = (u16*)(ws + 0);           // 6 MB
    u16*   wo   = (u16*)(ws + 6291456);     // 2 MB
    u16*   wff1 = (u16*)(ws + 8388608);     // 8 MB
    u16*   wff2 = (u16*)(ws + 16777216);    // 8 MB
    u16*   xn   = (u16*)(ws + 25165824);    // 16 MB
    u16*   qb   = (u16*)(ws + 41943040);    // 16 MB
    u16*   kb   = (u16*)(ws + 58720256);    // 16 MB
    u16*   vb   = (u16*)(ws + 75497472);    // 16 MB
    u16*   ctx  = (u16*)(ws + 25165824);    // reuses xn (dead after QKV gemm)
    float* attn = (float*)(ws + 41943040);  // 32 MB, reuses q+k (dead)
    u16*   yn   = (u16*)(ws + 75497472);    // reuses v (dead)
    u16*   hbuf = (u16*)(ws + 92274688);    // 64 MB; total ~152 MB

    const int M = 8192;  // 4*2048

    {
        int n;
        n = 1024 * 3072;
        f32_to_bf16_kernel<<<(n + 255) / 256, 256, 0, stream>>>(W_qkv, wqkv, n);
        n = 1024 * 1024;
        f32_to_bf16_kernel<<<(n + 255) / 256, 256, 0, stream>>>(W_o, wo, n);
        n = 1024 * 4096;
        f32_to_bf16_kernel<<<(n + 255) / 256, 256, 0, stream>>>(W_ff1, wff1, n);
        n = 4096 * 1024;
        f32_to_bf16_kernel<<<(n + 255) / 256, 256, 0, stream>>>(W_ff2, wff2, n);
    }

    rmsnorm_bf16_kernel<<<M, 256, 0, stream>>>(x, n1w, xn, 1024);

    gemm_bf16_kernel<2, false><<<dim3(3072 / 128, M / 128), 256, 0, stream>>>(
        xn, wqkv, b_qkv, nullptr, nullptr, qb, kb, vb, M, 3072, 1024);

    flash_attn_kernel<<<dim3(2048 / 64, 16, 4), 128, 0, stream>>>(qb, kb, vb, ctx);

    gemm_bf16_kernel<0, false><<<dim3(1024 / 128, M / 128), 256, 0, stream>>>(
        ctx, wo, b_o, attn, nullptr, nullptr, nullptr, nullptr, M, 1024, 1024);

    rmsnorm_bf16_kernel<<<M, 256, 0, stream>>>(attn, n2w, yn, 1024);

    gemm_bf16_kernel<1, true><<<dim3(4096 / 128, M / 128), 256, 0, stream>>>(
        yn, wff1, b_ff1, nullptr, hbuf, nullptr, nullptr, nullptr, M, 4096, 1024);

    gemm_bf16_kernel<0, false><<<dim3(1024 / 128, M / 128), 256, 0, stream>>>(
        hbuf, wff2, b_ff2, out, nullptr, nullptr, nullptr, nullptr, M, 1024, 4096);
}
